// ParallelBlockMultiHeadAttention_24498493456702
// MI455X (gfx1250) — matrix operand order repaired and checked by reading, not compiled, not run on hardware
//
#include <hip/hip_runtime.h>

#define B_   8
#define N_   1024
#define DIM_ 768
#define NH_  12
#define HD_  64

typedef _Float16 f16;
typedef __attribute__((ext_vector_type(16))) _Float16 v16h;
typedef __attribute__((ext_vector_type(8)))  float    v8f;
typedef unsigned int u32x4 __attribute__((ext_vector_type(4)));
typedef int          i32x8 __attribute__((ext_vector_type(8)));
typedef int          i32x4 __attribute__((ext_vector_type(4)));

__device__ __forceinline__ v8f wmma_f16(v16h a, v16h b, v8f c) {
  // D = A(16x32 f16) * B(32x16 f16) + C(16x16 f32)
  return __builtin_amdgcn_wmma_f32_16x16x32_f16(
      /*neg_a=*/false, a, /*neg_b=*/false, b,
      /*c_mod=*/(short)0, c, /*reuse_a=*/false, /*reuse_b=*/false);
}

// A-matrix 16x32 f16 fragment from row-major f16 tile in LDS.
// ISA layout: lane row = lane&15; element i holds k = (i&7) + (i>>3)*16 + (lane>>4)*8
__device__ __forceinline__ v16h load_A(const f16* sm, int stride, int kbase, int lane) {
  const int row = lane & 15, half = lane >> 4;
  const f16* base = sm + row * stride + kbase + half * 8;
  v16h a;
#pragma unroll
  for (int v = 0; v < 8; ++v) {
    const int k = (v & 3) * 2 + (v >> 2) * 16;  // pair-base for elements 2v,2v+1
    a[2 * v]     = base[k];
    a[2 * v + 1] = base[k + 1];
  }
  return a;
}

// B-matrix 32x16 f16 fragment. ISA layout: lane col = lane&15,
// element i holds k = (i&7) + (i>>3)*16 + (lane>>4)*8 (matches load_A).
// sm is laid out [col][k] row-major with given stride.
__device__ __forceinline__ v16h load_B(const f16* sm, int stride, int colbase,
                                       int kbase, int lane) {
  const int col = colbase + (lane & 15), half = lane >> 4;
  const f16* base = sm + col * stride + kbase + half * 8;
  v16h b;
#pragma unroll
  for (int i = 0; i < 16; ++i) b[i] = base[(i & 7) + (i >> 3) * 16];
  return b;
}

// Issue a TDM 2D tile load: 32 rows x 64 fp32, row stride = 3*DIM_ floats,
// Global -> LDS (row-major, contiguous). Tracked on TENSORcnt.
__device__ __forceinline__ void tdm_load_v_tile(const float* gsrc, void* lds_dst) {
  const unsigned long long ga = (unsigned long long)(uintptr_t)gsrc;
  const unsigned int lds_off = (unsigned int)(uintptr_t)lds_dst;  // generic addr low 32 = LDS offset

  u32x4 g0;
  g0.x = 1u;                                              // count=1 (valid user descriptor)
  g0.y = lds_off;                                         // lds_addr [63:32]
  g0.z = (unsigned int)(ga & 0xffffffffu);                // global_addr[31:0]  -> bits [95:64]
  g0.w = (unsigned int)((ga >> 32) & 0x1ffffffu)          // global_addr[56:32]
         | (2u << 30);                                    // type=2 ("image")   -> bits [127:126]

  i32x8 g1;
  g1[0] = (2 << 16);       // workgroup_mask=0, data_size=2 (4 bytes)
  g1[1] = (64 << 16);      // tensor_dim0 low16 = 64 (bits [63:48])
  g1[2] = (32 << 16);      // tensor_dim0 hi = 0; tensor_dim1 low16 = 32 (bits [95:80])
  g1[3] = (64 << 16);      // tensor_dim1 hi = 0; tile_dim0 = 64 (bits [127:112])
  g1[4] = 32;              // tile_dim1 = 32; tile_dim2 = 0
  g1[5] = 3 * DIM_;        // tensor_dim0_stride low32 = 2304 elements
  g1[6] = 0;               // stride hi, tensor_dim1_stride (unused for 2D)
  g1[7] = 0;

  i32x4 z4 = {0, 0, 0, 0};
#if defined(__clang_major__) && (__clang_major__ >= 23)
  i32x8 z8 = {0, 0, 0, 0, 0, 0, 0, 0};
  __builtin_amdgcn_tensor_load_to_lds(g0, g1, z4, z4, z8, 0);
#else
  __builtin_amdgcn_tensor_load_to_lds(g0, g1, z4, z4, 0);
#endif
}

// ---------------------------------------------------------------------------
// Kernel 1: fused per-head LayerNorm(Q), LayerNorm(K), flash attention.
// Grid: (N/128, NH, B), block 256 (8 waves). Wave w owns query rows
// [n0 + 16w, n0 + 16w + 16). Key chunks of 32 staged cooperatively in LDS;
// the V tile is fetched by the Tensor Data Mover while the VALU does K-LN.
// Writes x (B,N,DIM) fp32 to workspace.
// ---------------------------------------------------------------------------
__global__ __launch_bounds__(256) void attn_kernel(
    const float* __restrict__ QKV,
    const float* __restrict__ qw, const float* __restrict__ qb,
    const float* __restrict__ kw, const float* __restrict__ kb,
    float* __restrict__ X) {
  __shared__ f16   sQ[128 * 64];     // 8 waves x 16 rows x 64, layer-normed Q
  __shared__ f16   sK[32 * 64];      // 32 keys x 64, layer-normed K
  __shared__ f16   sVT[64 * 32];     // V transposed: [d][key], f16
  __shared__ f16   sP[8 * 16 * 32];  // per-wave softmax tile
  __shared__ float sVstage[32 * 64]; // raw fp32 V tile landed by the TDM

  const int tid  = threadIdx.x;
  const int wave = tid >> 5, lane = tid & 31;
  const int half = lane >> 4;
  const int b = blockIdx.z, h = blockIdx.y;
  const int n0 = blockIdx.x * 128;

  const size_t rs = 3 * DIM_;  // QKV row stride
  const float* qbase = QKV + (size_t)(b * N_) * rs + h * HD_;
  const float* kbase = qbase + DIM_;
  const float* vbase = qbase + 2 * DIM_;

  // ---- LayerNorm this wave's 16 Q rows into sQ (f16) ----
  {
    const float w0 = qw[2 * lane], w1 = qw[2 * lane + 1];
    const float b0 = qb[2 * lane], b1 = qb[2 * lane + 1];
    for (int r = 0; r < 16; ++r) {
      const int n = n0 + wave * 16 + r;
      const float* row = qbase + (size_t)n * rs;
      float x0 = row[2 * lane], x1 = row[2 * lane + 1];
      float s = x0 + x1, ss = x0 * x0 + x1 * x1;
#pragma unroll
      for (int m = 16; m >= 1; m >>= 1) {
        s  += __shfl_xor(s,  m, 32);
        ss += __shfl_xor(ss, m, 32);
      }
      const float mean = s * (1.f / 64.f);
      const float var  = ss * (1.f / 64.f) - mean * mean;
      const float rstd = rsqrtf(var + 1e-5f);
      sQ[(wave * 16 + r) * 64 + 2 * lane]     = (f16)((x0 - mean) * rstd * w0 + b0);
      sQ[(wave * 16 + r) * 64 + 2 * lane + 1] = (f16)((x1 - mean) * rstd * w1 + b1);
    }
  }
  __syncthreads();

  // Q fragments are loop-invariant
  const v16h aQ0 = load_A(sQ + wave * 16 * 64, 64, 0, lane);
  const v16h aQ1 = load_A(sQ + wave * 16 * 64, 64, 32, lane);

  v8f o0 = {}, o1 = {}, o2 = {}, o3 = {};
  float mrun[8], lrun[8];
#pragma unroll
  for (int j = 0; j < 8; ++j) { mrun[j] = -1e30f; lrun[j] = 0.f; }

  f16* sPw = sP + wave * 16 * 32;
  const float kw0 = kw[2 * lane], kw1 = kw[2 * lane + 1];
  const float kb0 = kb[2 * lane], kb1 = kb[2 * lane + 1];

  for (int ck = 0; ck < N_ / 32; ++ck) {
    __syncthreads();  // all waves done with previous sK/sVT/sVstage

    // ---- kick the TDM: V tile (32 x 64 fp32) -> sVstage, async ----
    if (tid == 0) {
      tdm_load_v_tile(vbase + (size_t)(ck * 32) * rs, sVstage);
    }

    // ---- stage K chunk with LayerNorm (wave handles 4 key rows) ----
    for (int kr = 0; kr < 4; ++kr) {
      const int key = wave * 4 + kr;
      const float* row = kbase + (size_t)(ck * 32 + key) * rs;
      float x0 = row[2 * lane], x1 = row[2 * lane + 1];
      float s = x0 + x1, ss = x0 * x0 + x1 * x1;
#pragma unroll
      for (int m = 16; m >= 1; m >>= 1) {
        s  += __shfl_xor(s,  m, 32);
        ss += __shfl_xor(ss, m, 32);
      }
      const float mean = s * (1.f / 64.f);
      const float var  = ss * (1.f / 64.f) - mean * mean;
      const float rstd = rsqrtf(var + 1e-5f);
      sK[key * 64 + 2 * lane]     = (f16)((x0 - mean) * rstd * kw0 + kb0);
      sK[key * 64 + 2 * lane + 1] = (f16)((x1 - mean) * rstd * kw1 + kb1);
    }
    if (ck + 1 < N_ / 32) {            // global_prefetch_b8 for next K chunk
      __builtin_prefetch(kbase + (size_t)((ck + 1) * 32 + wave * 4) * rs + lane, 0, 1);
    }

    // ---- drain the TDM on the issuing wave, then make it block-visible ----
    if (tid == 0) {
      __builtin_amdgcn_s_wait_tensorcnt(0);
    }
    __syncthreads();

    // ---- convert + transpose V: sVstage[key][d] fp32 -> sVT[d][key] f16 ----
#pragma unroll
    for (int i = 0; i < 8; ++i) {
      const int e = tid + i * 256;     // 2048 = 32 keys x 64 d
      const int key = e >> 6, d = e & 63;
      sVT[d * 32 + key] = (f16)sVstage[key * 64 + d];
    }
    __syncthreads();

    // ---- S = Q K^T : two 16x16 tiles over 32 keys, K-dim = 64 ----
    v8f s0 = {}, s1 = {};
    s0 = wmma_f16(aQ0, load_B(sK, 64, 0,  0,  lane), s0);
    s0 = wmma_f16(aQ1, load_B(sK, 64, 0,  32, lane), s0);
    s1 = wmma_f16(aQ0, load_B(sK, 64, 16, 0,  lane), s1);
    s1 = wmma_f16(aQ1, load_B(sK, 64, 16, 32, lane), s1);

    // ---- online softmax on C-fragment layout (row = j + half*8) ----
    const float scale = 0.125f;  // 64^-0.5
#pragma unroll
    for (int j = 0; j < 8; ++j) {
      float v0 = s0[j] * scale, v1 = s1[j] * scale;
      float mx = fmaxf(v0, v1);
#pragma unroll
      for (int m = 8; m >= 1; m >>= 1) mx = fmaxf(mx, __shfl_xor(mx, m, 16));
      const float mn = fmaxf(mrun[j], mx);
      const float p0 = __expf(v0 - mn), p1 = __expf(v1 - mn);
      const float corr = __expf(mrun[j] - mn);
      float rsum = p0 + p1;
#pragma unroll
      for (int m = 8; m >= 1; m >>= 1) rsum += __shfl_xor(rsum, m, 16);
      lrun[j] = lrun[j] * corr + rsum;
      mrun[j] = mn;
      o0[j] *= corr; o1[j] *= corr; o2[j] *= corr; o3[j] *= corr;
      const int r = j + half * 8;
      sPw[r * 32 + (lane & 15)]      = (f16)p0;
      sPw[r * 32 + 16 + (lane & 15)] = (f16)p1;
    }

    // ---- O += P V : P is 16x32 A-fragment, V^T gives contiguous B-frags ----
    const v16h aP = load_A(sPw, 32, 0, lane);
    o0 = wmma_f16(aP, load_B(sVT, 32, 0,  0, lane), o0);
    o1 = wmma_f16(aP, load_B(sVT, 32, 16, 0, lane), o1);
    o2 = wmma_f16(aP, load_B(sVT, 32, 32, 0, lane), o2);
    o3 = wmma_f16(aP, load_B(sVT, 32, 48, 0, lane), o3);
  }

  // ---- finalize: x[b][n][h*64+d] = O / l ----
#pragma unroll
  for (int j = 0; j < 8; ++j) {
    const int n = n0 + wave * 16 + j + half * 8;
    const float inv = 1.f / lrun[j];
    float* out = X + ((size_t)(b * N_ + n)) * DIM_ + h * HD_ + (lane & 15);
    out[0]  = o0[j] * inv;
    out[16] = o1[j] * inv;
    out[32] = o2[j] * inv;
    out[48] = o3[j] * inv;
  }
}

// ---------------------------------------------------------------------------
// Kernel 2: out = x @ W^T + bias.  x:(8192,768) f32, W:(768,768) f32 row-major.
// One wave = 16x16 output tile; block = 8 waves covering 32(m) x 64(o).
// Operands converted f32->f16 on the fly; W stays hot in the 192MB L2.
// ---------------------------------------------------------------------------
__global__ __launch_bounds__(256) void proj_kernel(
    const float* __restrict__ X, const float* __restrict__ W,
    const float* __restrict__ bias, float* __restrict__ out) {
  const int tid  = threadIdx.x;
  const int wave = tid >> 5, lane = tid & 31;
  const int half = lane >> 4;
  const int otile = blockIdx.x * 64 + (wave & 3) * 16;
  const int mtile = blockIdx.y * 32 + (wave >> 2) * 16;

  const int row = mtile + (lane & 15);  // A-frag row
  const int col = otile + (lane & 15);  // B-frag col / C col
  const float* xrow = X + (size_t)row * DIM_;
  const float* wrow = W + (size_t)col * DIM_;

  v8f acc = {};
  for (int cc = 0; cc < DIM_; cc += 32) {
    v16h a, bf;
    const float* ax = xrow + cc + half * 8;
#pragma unroll
    for (int v = 0; v < 8; ++v) {
      const int k = (v & 3) * 2 + (v >> 2) * 16;
      a[2 * v]     = (f16)ax[k];
      a[2 * v + 1] = (f16)ax[k + 1];
    }
    const float* bx = wrow + cc + half * 8;
#pragma unroll
    for (int i = 0; i < 16; ++i) bf[i] = (f16)bx[(i & 7) + (i >> 3) * 16];
    acc = wmma_f16(a, bf, acc);
  }

  const float bv = bias[col];
#pragma unroll
  for (int j = 0; j < 8; ++j) {
    const int mrow = mtile + j + half * 8;
    out[(size_t)mrow * DIM_ + col] = acc[j] + bv;
  }
}

extern "C" void kernel_launch(void* const* d_in, const int* in_sizes, int n_in,
                              void* d_out, int out_size, void* d_ws, size_t ws_size,
                              hipStream_t stream) {
  const float* QKV = (const float*)d_in[0];
  const float* qw  = (const float*)d_in[1];
  const float* qb  = (const float*)d_in[2];
  const float* kw  = (const float*)d_in[3];
  const float* kb  = (const float*)d_in[4];
  const float* W   = (const float*)d_in[5];
  const float* pb  = (const float*)d_in[6];
  float* X = (float*)d_ws;  // (B, N, DIM) fp32 attention output

  attn_kernel<<<dim3(N_ / 128, NH_, B_), 256, 0, stream>>>(QKV, qw, qb, kw, kb, X);
  proj_kernel<<<dim3(DIM_ / 64, (B_ * N_) / 32), 256, 0, stream>>>(X, W, pb, (float*)d_out);
}
